// GPT_41051297415311
// MI455X (gfx1250) — matrix operand order repaired and checked by reading, not compiled, not run on hardware
//
#include <hip/hip_runtime.h>
#include <math.h>

// ---------------------------------------------------------------------------
// GPT-2 small forward on gfx1250 (CDNA5): all GEMMs + attention on
// v_wmma_f32_16x16x32_f16 (f16 inputs, f32 accumulate).
// ---------------------------------------------------------------------------

typedef __attribute__((ext_vector_type(16))) _Float16 v16h;
typedef __attribute__((ext_vector_type(8)))  float    v8f;

union AFrag { v16h v; uint4 q[2]; _Float16 h[16]; };
union CFrag { v8f  v; float f[8]; };

__device__ __forceinline__ v8f wmma_f16(v16h a, v16h b, v8f c) {
  // D = A(16x32) * B(32x16) + C, f32 accumulate
  return __builtin_amdgcn_wmma_f32_16x16x32_f16(
      /*neg_a=*/false, a, /*neg_b=*/false, b,
      /*c_mod=*/(short)0, c, /*reuse_a=*/false, /*reuse_b=*/false);
}

// ---------------------------------------------------------------------------
// f32 -> f16 weight conversion (grid-stride)
// ---------------------------------------------------------------------------
__global__ __launch_bounds__(256) void f32_to_f16_kernel(
    const float* __restrict__ s, _Float16* __restrict__ d, size_t n) {
  size_t i = (size_t)blockIdx.x * blockDim.x + threadIdx.x;
  size_t stride = (size_t)gridDim.x * blockDim.x;
  for (; i < n; i += stride) d[i] = (_Float16)s[i];
}

// ---------------------------------------------------------------------------
// Embedding: x[b,t,:] = wte[idx[b,t],:] + wpe[t,:]
// ---------------------------------------------------------------------------
__global__ __launch_bounds__(256) void embed_kernel(
    const int* __restrict__ idx, const float* __restrict__ wte,
    const float* __restrict__ wpe, float* __restrict__ X) {
  const int bt = blockIdx.x;           // 0..B*T-1
  const int t = bt & 1023;             // T = 1024
  const int tok = idx[bt];
  for (int j = threadIdx.x; j < 768; j += 256)
    X[(size_t)bt * 768 + j] = wte[(size_t)tok * 768 + j] + wpe[(size_t)t * 768 + j];
}

// ---------------------------------------------------------------------------
// LayerNorm over D=768, one workgroup (256 threads) per row; f16 output.
// ---------------------------------------------------------------------------
__global__ __launch_bounds__(256) void ln_kernel(
    const float* __restrict__ X, const float* __restrict__ Wg,
    const float* __restrict__ Bg, _Float16* __restrict__ Out) {
  __shared__ float red[16];
  const int tid = threadIdx.x;
  const size_t row = blockIdx.x;
  const float* xr = X + row * 768;
  float v0 = xr[tid], v1 = xr[tid + 256], v2 = xr[tid + 512];
  float s = v0 + v1 + v2;
  for (int o = 1; o < 32; o <<= 1) s += __shfl_xor(s, o);
  if ((tid & 31) == 0) red[tid >> 5] = s;
  __syncthreads();
  float tot = 0.0f;
#pragma unroll
  for (int i = 0; i < 8; ++i) tot += red[i];
  const float mean = tot * (1.0f / 768.0f);
  const float d0 = v0 - mean, d1 = v1 - mean, d2 = v2 - mean;
  float s2 = d0 * d0 + d1 * d1 + d2 * d2;
  for (int o = 1; o < 32; o <<= 1) s2 += __shfl_xor(s2, o);
  if ((tid & 31) == 0) red[8 + (tid >> 5)] = s2;
  __syncthreads();
  float tot2 = 0.0f;
#pragma unroll
  for (int i = 0; i < 8; ++i) tot2 += red[8 + i];
  const float rstd = rsqrtf(tot2 * (1.0f / 768.0f) + 1e-5f);
  Out[row * 768 + tid]       = (_Float16)(d0 * rstd * Wg[tid] + Bg[tid]);
  Out[row * 768 + tid + 256] = (_Float16)(d1 * rstd * Wg[tid + 256] + Bg[tid + 256]);
  Out[row * 768 + tid + 512] = (_Float16)(d2 * rstd * Wg[tid + 512] + Bg[tid + 512]);
}

// ---------------------------------------------------------------------------
// Per-head V transpose: Vt[b][h][d][t] = V[b][t][h*64+d]  (f16)
// Grid: (H, B, 8); each block covers 128 t values.
// ---------------------------------------------------------------------------
__global__ __launch_bounds__(256) void transpose_v_kernel(
    const _Float16* __restrict__ Vh, _Float16* __restrict__ Vt) {
  const int h = blockIdx.x, bz = blockIdx.y;
  const int d = threadIdx.x & 63;
  const int tl = threadIdx.x >> 6;  // 0..3
  const size_t src0 = ((size_t)bz * 1024) * 768 + h * 64 + d;
  const size_t dst0 = ((size_t)((bz * 12 + h) * 64 + d)) * 1024;
  const int tb = blockIdx.z * 128 + tl * 32;
#pragma unroll 8
  for (int tt = 0; tt < 32; ++tt) {
    const int t = tb + tt;
    Vt[dst0 + t] = Vh[src0 + (size_t)t * 768];
  }
}

// ---------------------------------------------------------------------------
// WMMA GEMM: C[M,N] = A[M,K] @ B + bias (+resid) (+gelu)
//   BNK = 0: B is [K,N] row-major (N % 64 == 0 assumed)
//   BNK = 1: B is [N,K] row-major (i.e. C = A @ B^T), N edge guarded
//   EPI = 0: f32 out        EPI = 1: f16 out
//   EPI = 2: gelu, f16 out  EPI = 3: resid add, f32 out
// Block tile 128x64, BK=32, 8 waves laid out 4x2; each wave: 32x32 output
// (2 A frags x 2 B frags -> 4 WMMAs per K-step).  M % 128 == 0 assumed.
// ---------------------------------------------------------------------------
template <int BNK, int EPI>
__global__ __launch_bounds__(256) void wmma_gemm_kernel(
    const _Float16* __restrict__ A, const _Float16* __restrict__ Bm,
    const float* __restrict__ bias, const float* __restrict__ resid,
    float* __restrict__ Cf, _Float16* __restrict__ Ch,
    int M, int N, int K) {
  __shared__ _Float16 As[128 * 32];  // [m][k]
  __shared__ _Float16 Bt[64 * 32];   // [n][k]  (B staged transposed)
  const int tid = threadIdx.x;
  const int lane = tid & 31, w = tid >> 5;
  const int g = lane >> 4, ln = lane & 15;
  const int wr = w >> 1, wc = w & 1;       // wave grid 4 x 2
  const int m0 = blockIdx.y * 128, n0 = blockIdx.x * 64;

  CFrag acc[2][2];
#pragma unroll
  for (int i = 0; i < 2; ++i)
#pragma unroll
    for (int j = 0; j < 2; ++j)
#pragma unroll
      for (int r = 0; r < 8; ++r) acc[i][j].f[r] = 0.0f;

  for (int k0 = 0; k0 < K; k0 += 32) {
    {  // stage A tile: 128 rows x 32 halves; 32B per thread, coalesced
      const int row = tid >> 1, seg = tid & 1;
      const _Float16* ap = &A[(size_t)(m0 + row) * K + k0 + seg * 16];
      *(uint4*)&As[row * 32 + seg * 16]     = *(const uint4*)ap;
      *(uint4*)&As[row * 32 + seg * 16 + 8] = *(const uint4*)(ap + 8);
      if (k0 + 32 < K) __builtin_prefetch(ap + 32, 0, 1);  // global_prefetch_b8
    }
    if (BNK) {  // B is [N,K]: rows are already the transposed layout
      const int row = tid >> 2, seg = tid & 3;
      uint4 val; val.x = val.y = val.z = val.w = 0u;
      if (n0 + row < N)
        val = *(const uint4*)&Bm[(size_t)(n0 + row) * K + k0 + seg * 8];
      *(uint4*)&Bt[row * 32 + seg * 8] = val;
    } else {    // B is [K,N]: load rows coalesced, scatter transposed to LDS
      const int kk = tid >> 3, nseg = tid & 7;
      union { uint4 q; _Float16 h[8]; } tmp;
      tmp.q = *(const uint4*)&Bm[(size_t)(k0 + kk) * N + n0 + nseg * 8];
#pragma unroll
      for (int i2 = 0; i2 < 8; ++i2)
        Bt[(nseg * 8 + i2) * 32 + kk] = tmp.h[i2];
    }
    __syncthreads();

    // A fragment: lane(g, m=ln): e=0..7 -> k=8g+e ; e=8..15 -> k=16+8g+(e-8)
    AFrag a0, a1, b0, b1;
    const int ar0 = (wr * 32 + ln) * 32;
    a0.q[0] = *(const uint4*)&As[ar0 + 8 * g];
    a0.q[1] = *(const uint4*)&As[ar0 + 16 + 8 * g];
    const int ar1 = (wr * 32 + 16 + ln) * 32;
    a1.q[0] = *(const uint4*)&As[ar1 + 8 * g];
    a1.q[1] = *(const uint4*)&As[ar1 + 16 + 8 * g];
    // B fragment: lane(g, n=ln): e=0..7 -> k=8g+e ; e=8..15 -> k=16+8g+(e-8)
    const int br0 = (wc * 32 + ln) * 32;
    b0.q[0] = *(const uint4*)&Bt[br0 + 8 * g];
    b0.q[1] = *(const uint4*)&Bt[br0 + 16 + 8 * g];
    const int br1 = (wc * 32 + 16 + ln) * 32;
    b1.q[0] = *(const uint4*)&Bt[br1 + 8 * g];
    b1.q[1] = *(const uint4*)&Bt[br1 + 16 + 8 * g];

    acc[0][0].v = wmma_f16(a0.v, b0.v, acc[0][0].v);
    acc[0][1].v = wmma_f16(a0.v, b1.v, acc[0][1].v);
    acc[1][0].v = wmma_f16(a1.v, b0.v, acc[1][0].v);
    acc[1][1].v = wmma_f16(a1.v, b1.v, acc[1][1].v);
    __syncthreads();
  }

  // epilogue; C layout: VGPR r -> row (r + 8g), col = lane&15
#pragma unroll
  for (int i = 0; i < 2; ++i) {
#pragma unroll
    for (int j = 0; j < 2; ++j) {
      const int col = n0 + wc * 32 + j * 16 + ln;
      if (col < N) {
        const float bv = bias ? bias[col] : 0.0f;
#pragma unroll
        for (int r = 0; r < 8; ++r) {
          const size_t rowm = (size_t)(m0 + wr * 32 + i * 16 + r + 8 * g);
          float val = acc[i][j].f[r] + bv;
          if (EPI == 3) val += resid[rowm * (size_t)N + col];
          if (EPI == 2) val = 0.5f * val * (1.0f + erff(val * 0.70710678118f));
          if (EPI == 0 || EPI == 3) Cf[rowm * (size_t)N + col] = val;
          else                      Ch[rowm * (size_t)N + col] = (_Float16)val;
        }
      }
    }
  }
}

// ---------------------------------------------------------------------------
// Flash-style causal attention. Grid: (T/128, H, B), 256 threads (8 waves).
// Wave w handles query rows [qb, qb+16), qb = bx*128 + w*16.  HD = 64.
// V is pre-transposed: Vt[b][h][d][t] so PV B-fragments are contiguous.
// ---------------------------------------------------------------------------
__global__ __launch_bounds__(256) void attn_kernel(
    const _Float16* __restrict__ Q, const _Float16* __restrict__ Km,
    const _Float16* __restrict__ Vt, _Float16* __restrict__ Y) {
  __shared__ _Float16 Plds[8][16][32];  // per-wave P tile (16 q x 32 keys)
  const int tid = threadIdx.x;
  const int w = tid >> 5, lane = tid & 31;
  const int g = lane >> 4, ln = lane & 15;
  const int qb = blockIdx.x * 128 + w * 16;
  const int bz = blockIdx.z;
  const int hoff = blockIdx.y * 64;
  const size_t vbase = ((size_t)((bz * 12 + blockIdx.y) * 64)) * 1024;

  // Q tile as two A fragments (K-steps d0 = 0, 32 over HD=64)
  const size_t rowQ = ((size_t)(bz * 1024 + qb + ln)) * 768 + hoff;
  AFrag qa0, qa1;
  qa0.q[0] = *(const uint4*)(Q + rowQ + 8 * g);
  qa0.q[1] = *(const uint4*)(Q + rowQ + 16 + 8 * g);
  qa1.q[0] = *(const uint4*)(Q + rowQ + 32 + 8 * g);
  qa1.q[1] = *(const uint4*)(Q + rowQ + 48 + 8 * g);

  float m_run[8], l_run[8];
  CFrag o[4];
#pragma unroll
  for (int r = 0; r < 8; ++r) { m_run[r] = -1e30f; l_run[r] = 0.0f; }
#pragma unroll
  for (int d = 0; d < 4; ++d)
#pragma unroll
    for (int r = 0; r < 8; ++r) o[d].f[r] = 0.0f;

  for (int t0 = 0; t0 < qb + 16; t0 += 32) {  // wave-uniform bound
    CFrag s0, s1;
#pragma unroll
    for (int r = 0; r < 8; ++r) { s0.f[r] = 0.0f; s1.f[r] = 0.0f; }
    {  // keys t0 .. t0+15
      const size_t rk = ((size_t)(bz * 1024 + t0 + ln)) * 768 + hoff;
      AFrag kb;
      kb.q[0] = *(const uint4*)(Km + rk + 8 * g);
      kb.q[1] = *(const uint4*)(Km + rk + 16 + 8 * g);
      s0.v = wmma_f16(qa0.v, kb.v, s0.v);
      kb.q[0] = *(const uint4*)(Km + rk + 32 + 8 * g);
      kb.q[1] = *(const uint4*)(Km + rk + 48 + 8 * g);
      s0.v = wmma_f16(qa1.v, kb.v, s0.v);
    }
    {  // keys t0+16 .. t0+31
      const size_t rk = ((size_t)(bz * 1024 + t0 + 16 + ln)) * 768 + hoff;
      AFrag kb;
      kb.q[0] = *(const uint4*)(Km + rk + 8 * g);
      kb.q[1] = *(const uint4*)(Km + rk + 16 + 8 * g);
      s1.v = wmma_f16(qa0.v, kb.v, s1.v);
      kb.q[0] = *(const uint4*)(Km + rk + 32 + 8 * g);
      kb.q[1] = *(const uint4*)(Km + rk + 48 + 8 * g);
      s1.v = wmma_f16(qa1.v, kb.v, s1.v);
    }
    // online softmax (row = r + 8g across 16 lanes of the half-wave)
    float p0[8], p1[8];
#pragma unroll
    for (int r = 0; r < 8; ++r) {
      const int qi = qb + r + 8 * g;
      float a0 = s0.f[r] * 0.125f;  // 1/sqrt(64)
      float a1 = s1.f[r] * 0.125f;
      if (t0 + ln > qi)      a0 = -1e30f;
      if (t0 + 16 + ln > qi) a1 = -1e30f;
      float tm = fmaxf(a0, a1);
      tm = fmaxf(tm, __shfl_xor(tm, 1));
      tm = fmaxf(tm, __shfl_xor(tm, 2));
      tm = fmaxf(tm, __shfl_xor(tm, 4));
      tm = fmaxf(tm, __shfl_xor(tm, 8));
      const float mn = fmaxf(m_run[r], tm);
      const float sc = __expf(m_run[r] - mn);
      const float e0 = __expf(a0 - mn), e1 = __expf(a1 - mn);
      float rs = e0 + e1;
      rs += __shfl_xor(rs, 1);
      rs += __shfl_xor(rs, 2);
      rs += __shfl_xor(rs, 4);
      rs += __shfl_xor(rs, 8);
      l_run[r] = l_run[r] * sc + rs;
      m_run[r] = mn;
#pragma unroll
      for (int d = 0; d < 4; ++d) o[d].f[r] *= sc;
      p0[r] = e0; p1[r] = e1;
    }
    // C-layout -> A-layout via per-wave LDS (in-order DS, wave-private)
#pragma unroll
    for (int r = 0; r < 8; ++r) {
      Plds[w][r + 8 * g][ln]      = (_Float16)p0[r];
      Plds[w][r + 8 * g][16 + ln] = (_Float16)p1[r];
    }
    AFrag pa;
    pa.q[0] = *(const uint4*)&Plds[w][ln][8 * g];
    pa.q[1] = *(const uint4*)&Plds[w][ln][16 + 8 * g];
    // O += P @ V : B fragment e<8 -> key kk = 8g+e, e>=8 -> 16+8g+(e-8), n = d; Vt rows contiguous
#pragma unroll
    for (int d = 0; d < 4; ++d) {
      const _Float16* vp = Vt + vbase + (size_t)(d * 16 + ln) * 1024 + t0 + 8 * g;
      AFrag vbf;
      vbf.q[0] = *(const uint4*)vp;
      vbf.q[1] = *(const uint4*)(vp + 16);
      o[d].v = wmma_f16(pa.v, vbf.v, o[d].v);
    }
  }
  // finalize: divide by l, write f16 y[b,t, h*64 + d]
#pragma unroll
  for (int d = 0; d < 4; ++d)
#pragma unroll
    for (int r = 0; r < 8; ++r)
      Y[((size_t)(bz * 1024 + qb + r + 8 * g)) * 768 + hoff + d * 16 + ln] =
          (_Float16)(o[d].f[r] / l_run[r]);
}

// ---------------------------------------------------------------------------
// Host orchestration
// ---------------------------------------------------------------------------
extern "C" void kernel_launch(void* const* d_in, const int* in_sizes, int n_in,
                              void* d_out, int out_size, void* d_ws, size_t ws_size,
                              hipStream_t stream) {
  (void)in_sizes; (void)n_in; (void)out_size; (void)ws_size;
  constexpr int Lc = 4, Dm = 768, Tc = 1024, Vc = 50257, Bc = 2, Hc = 12;
  constexpr int Dff = 4 * Dm;
  constexpr int BT = Bc * Tc;  // 2048

  const int*   idx   = (const int*)  d_in[0];
  const float* wte   = (const float*)d_in[1];
  const float* wpe   = (const float*)d_in[2];
  const float* ln1w  = (const float*)d_in[3];
  const float* ln1b  = (const float*)d_in[4];
  const float* Wq    = (const float*)d_in[5];
  const float* bq    = (const float*)d_in[6];
  const float* Wk    = (const float*)d_in[7];
  const float* bk    = (const float*)d_in[8];
  const float* Wv    = (const float*)d_in[9];
  const float* bv    = (const float*)d_in[10];
  const float* Wo    = (const float*)d_in[11];
  const float* bo    = (const float*)d_in[12];
  const float* ln2w  = (const float*)d_in[13];
  const float* ln2b  = (const float*)d_in[14];
  const float* Wfc   = (const float*)d_in[15];
  const float* bfc   = (const float*)d_in[16];
  const float* Wpr   = (const float*)d_in[17];
  const float* bpr   = (const float*)d_in[18];
  const float* lnfw  = (const float*)d_in[19];
  const float* lnfb  = (const float*)d_in[20];
  const float* headw = (const float*)d_in[21];

  char* ws = (char*)d_ws;
  size_t off = 0;
  auto alloc = [&](size_t bytes) -> void* {
    void* p = ws + off;
    off += (bytes + 255) & ~(size_t)255;
    return p;
  };
  float*    x    = (float*)   alloc((size_t)BT * Dm * 4);
  _Float16* xn   = (_Float16*)alloc((size_t)BT * Dm * 2);
  _Float16* qh   = (_Float16*)alloc((size_t)BT * Dm * 2);
  _Float16* kh   = (_Float16*)alloc((size_t)BT * Dm * 2);
  _Float16* vh   = (_Float16*)alloc((size_t)BT * Dm * 2);
  _Float16* vt   = (_Float16*)alloc((size_t)BT * Dm * 2);  // [B,H,64,T]
  _Float16* yh   = (_Float16*)alloc((size_t)BT * Dm * 2);
  _Float16* hh   = (_Float16*)alloc((size_t)BT * Dff * 2);
  _Float16* wqh  = (_Float16*)alloc((size_t)Lc * Dm * Dm * 2);
  _Float16* wkh  = (_Float16*)alloc((size_t)Lc * Dm * Dm * 2);
  _Float16* wvh  = (_Float16*)alloc((size_t)Lc * Dm * Dm * 2);
  _Float16* woh  = (_Float16*)alloc((size_t)Lc * Dm * Dm * 2);
  _Float16* wfch = (_Float16*)alloc((size_t)Lc * Dm * Dff * 2);
  _Float16* wprh = (_Float16*)alloc((size_t)Lc * Dff * Dm * 2);
  _Float16* whd  = (_Float16*)alloc((size_t)Vc * Dm * 2);

  auto conv = [&](const float* s, _Float16* d, size_t n) {
    f32_to_f16_kernel<<<2048, 256, 0, stream>>>(s, d, n);
  };
  conv(Wq,    wqh,  (size_t)Lc * Dm * Dm);
  conv(Wk,    wkh,  (size_t)Lc * Dm * Dm);
  conv(Wv,    wvh,  (size_t)Lc * Dm * Dm);
  conv(Wo,    woh,  (size_t)Lc * Dm * Dm);
  conv(Wfc,   wfch, (size_t)Lc * Dm * Dff);
  conv(Wpr,   wprh, (size_t)Lc * Dff * Dm);
  conv(headw, whd,  (size_t)Vc * Dm);

  embed_kernel<<<BT, 256, 0, stream>>>(idx, wte, wpe, x);

  const dim3 gD(Dm / 64, BT / 128);     // 12 x 16
  const dim3 gF(Dff / 64, BT / 128);    // 48 x 16
  for (int l = 0; l < Lc; ++l) {
    ln_kernel<<<BT, 256, 0, stream>>>(x, ln1w + l * Dm, ln1b + l * Dm, xn);
    wmma_gemm_kernel<0, 1><<<gD, 256, 0, stream>>>(
        xn, wqh + (size_t)l * Dm * Dm, bq + l * Dm, nullptr, nullptr, qh, BT, Dm, Dm);
    wmma_gemm_kernel<0, 1><<<gD, 256, 0, stream>>>(
        xn, wkh + (size_t)l * Dm * Dm, bk + l * Dm, nullptr, nullptr, kh, BT, Dm, Dm);
    wmma_gemm_kernel<0, 1><<<gD, 256, 0, stream>>>(
        xn, wvh + (size_t)l * Dm * Dm, bv + l * Dm, nullptr, nullptr, vh, BT, Dm, Dm);
    transpose_v_kernel<<<dim3(Hc, Bc, 8), 256, 0, stream>>>(vh, vt);
    attn_kernel<<<dim3(Tc / 128, Hc, Bc), 256, 0, stream>>>(qh, kh, vt, yh);
    wmma_gemm_kernel<0, 3><<<gD, 256, 0, stream>>>(
        yh, woh + (size_t)l * Dm * Dm, bo + l * Dm, x, x, nullptr, BT, Dm, Dm);
    ln_kernel<<<BT, 256, 0, stream>>>(x, ln2w + l * Dm, ln2b + l * Dm, xn);
    wmma_gemm_kernel<0, 2><<<gF, 256, 0, stream>>>(
        xn, wfch + (size_t)l * Dm * Dff, bfc + l * Dff, nullptr, nullptr, hh, BT, Dff, Dm);
    wmma_gemm_kernel<0, 3><<<gD, 256, 0, stream>>>(
        hh, wprh + (size_t)l * Dff * Dm, bpr + l * Dm, x, x, nullptr, BT, Dm, Dff);
  }
  ln_kernel<<<BT, 256, 0, stream>>>(x, lnfw, lnfb, xn);
  // logits = xn @ head_w^T  (head_w is [V,D] row-major -> BNK=1 path)
  wmma_gemm_kernel<1, 0><<<dim3((Vc + 63) / 64, BT / 128), 256, 0, stream>>>(
      xn, whd, nullptr, nullptr, (float*)d_out, nullptr, BT, Vc, Dm);
}
